// MergerNNet_10823317585953
// MI455X (gfx1250) — matrix operand order repaired and checked by reading, not compiled, not run on hardware
//
#include <hip/hip_runtime.h>
#include <hip/hip_bf16.h>
#include <math.h>

typedef __attribute__((ext_vector_type(16))) _Float16 v16h;
typedef __attribute__((ext_vector_type(8)))  float    v8f;

#define WPB 4     // waves per block
#define TB  128   // threads per block (4 x wave32)

// ---------------------------------------------------------------------------
// Wave-local LDS fence: all prior DS/flat-LDS traffic complete + sched barrier.
// CDNA5 split counters: s_wait_dscnt / s_wait_storecnt.
// ---------------------------------------------------------------------------
__device__ __forceinline__ void wave_dsync() {
  asm volatile("s_wait_dscnt 0\n\ts_wait_storecnt 0" ::: "memory");
  __builtin_amdgcn_wave_barrier();
}

// ---------------------------------------------------------------------------
// A-fragment (16x32 f16) from LDS row-major tile [16][lda].
// Layout (matches B swizzle): lanes0-15 row M=lane: halves0-15=K+0..15
//             lanes16-31 row M=lane-16: halves0-15=K+16..31
// ---------------------------------------------------------------------------
__device__ __forceinline__ v16h load_A(const _Float16* tile, int lda, int k0, int lane) {
  int r = lane & 15;
  const _Float16* p = tile + r * lda + k0 + ((lane & 16) ? 16 : 0);
  union { v16h v; uint4 q[2]; } u;
  u.q[0] = *(const uint4*)(p);        // K + 0..7   (of this half-wave's base)
  u.q[1] = *(const uint4*)(p + 8);    // K + 8..15
  return u.v;
}

// ---------------------------------------------------------------------------
// B-fragment (32x16 f16) from pre-swizzled weights: one contiguous 32B load.
// Swizzle layout: dst[(((k0/32)*64 + n)*2 + halfwave)*16 + i] = W[k][n]
//   with k = (k0/32)*32 + halfwave*16 + i   (lanes0-15: K 0..15, 16-31: 16..31)
// ---------------------------------------------------------------------------
__device__ __forceinline__ v16h load_B(const _Float16* Wswz, int n0, int k0, int lane) {
  int n = n0 + (lane & 15);
  int h = (lane & 16) ? 1 : 0;
  int kb = k0 >> 5;
  return *(const v16h*)(Wswz + ((((kb << 6) + n) * 2 + h) << 4));
}

// ---------------------------------------------------------------------------
// One dense layer: out[16][64] (f32, LDS) = in[16][K] (f16, LDS) @ W[K][64] + b
// ---------------------------------------------------------------------------
__device__ __forceinline__ void wmma_layer(const _Float16* inT, int lda, int K,
                                           float* outT, const _Float16* Wswz,
                                           const float* bias, int lane) {
#pragma unroll
  for (int nt = 0; nt < 4; ++nt) {
    int n0 = nt << 4;
    float bv = bias[n0 + (lane & 15)];
    v8f acc;
#pragma unroll
    for (int i = 0; i < 8; ++i) acc[i] = bv;
    for (int k0 = 0; k0 < K; k0 += 32) {
      v16h a = load_A(inT, lda, k0, lane);
      v16h b = load_B(Wswz, n0, k0, lane);
      acc = __builtin_amdgcn_wmma_f32_16x16x32_f16(false, a, false, b,
                                                   (short)0, acc, false, false);
    }
    // C/D layout: VGPR v -> row v (lanes0-15) / v+8 (lanes16-31), col n0+lane%16
    int r  = lane & 15;
    int mb = (lane & 16) ? 8 : 0;
#pragma unroll
    for (int v = 0; v < 8; ++v) outT[(mb + v) * 64 + n0 + r] = acc[v];
  }
}

// LayerNorm(64) + ReLU, f32 LDS rows -> f16 LDS rows (lanes 0-15 own one row each)
__device__ __forceinline__ void ln_relu_f16(const float* outT, _Float16* dst, int ldd,
                                            const float* g, const float* bt, int lane) {
  wave_dsync();
  if (lane < 16) {
    const float* row = outT + lane * 64;
    float m = 0.f;
    for (int j = 0; j < 64; ++j) m += row[j];
    m *= (1.0f / 64.0f);
    float var = 0.f;
    for (int j = 0; j < 64; ++j) { float d = row[j] - m; var += d * d; }
    var *= (1.0f / 64.0f);
    float inv = 1.0f / sqrtf(var + 1e-5f);
    _Float16* drow = dst + lane * ldd;
    for (int j = 0; j < 64; ++j) {
      float y = (row[j] - m) * inv * g[j] + bt[j];
      drow[j] = (_Float16)(y > 0.f ? y : 0.f);
    }
  }
  wave_dsync();
}

// ---------------------------------------------------------------------------
// Edge MLP (edge_net / edge_dec): per wave, 16 edges through 128->64->64->64->1
// ---------------------------------------------------------------------------
__global__ __launch_bounds__(TB) void edge_mlp_kernel(
    const _Float16* __restrict__ xh,
    const _Float16* W1, const _Float16* W2, const _Float16* W3,
    const float* b1, const float* g1, const float* t1,
    const float* b2, const float* g2, const float* t2,
    const float* b3, const float* g3, const float* t3,
    const float* W4, const float* b4,
    float* __restrict__ eout, int NVT, int NCT, int E) {
  __shared__ __align__(16) _Float16 sIn[WPB][16 * 128];
  __shared__ __align__(16) float    sOut[WPB][16 * 64];
  int wave = threadIdx.x >> 5, lane = threadIdx.x & 31;
  int tile = blockIdx.x * WPB + wave;
  int e0 = tile << 4;
  if (e0 >= E) return;                       // wave-uniform exit
  _Float16* inT = sIn[wave];
  float*    outT = sOut[wave];

  // gather edge_in = [x[start] , x[end]] : 16 rows x 128 f16 (8 uint4/lane)
  for (int c = lane; c < 256; c += 32) {
    int row = c >> 4, part = c & 15;
    int eidx = e0 + row; if (eidx >= E) eidx = E - 1;
    int sN = eidx / NCT;
    int eN = eidx - sN * NCT + NVT;
    int node = (part < 8) ? sN : eN;
    int col = (part & 7) << 3;
    *(uint4*)(inT + (row << 7) + (part << 3)) = *(const uint4*)(xh + (node << 6) + col);
  }
  wave_dsync();

  wmma_layer(inT, 128, 128, outT, W1, b1, lane);
  ln_relu_f16(outT, inT, 64, g1, t1, lane);
  wmma_layer(inT, 64, 64, outT, W2, b2, lane);
  ln_relu_f16(outT, inT, 64, g2, t2, lane);
  wmma_layer(inT, 64, 64, outT, W3, b3, lane);
  ln_relu_f16(outT, inT, 64, g3, t3, lane);

  int eidx = e0 + lane;
  if (lane < 16 && eidx < E) {               // 64 -> 1, sigmoid
    float acc = b4[0];
    const _Float16* row = inT + (lane << 6);
    for (int j = 0; j < 64; ++j) acc += (float)row[j] * W4[j];
    eout[eidx] = 1.0f / (1.0f + expf(-acc));
  }
}

// ---------------------------------------------------------------------------
// Node MLP (node_net): [mi,mo,x] 192 -> 64 x4 layers (all LN+ReLU) + residual
// ---------------------------------------------------------------------------
__global__ __launch_bounds__(TB) void node_mlp_kernel(
    const float* __restrict__ mi, const float* __restrict__ mo,
    float* __restrict__ x, _Float16* __restrict__ xh,
    const _Float16* W1, const _Float16* W2, const _Float16* W3, const _Float16* W4s,
    const float* b1, const float* g1, const float* t1,
    const float* b2, const float* g2, const float* t2,
    const float* b3, const float* g3, const float* t3,
    const float* b4, const float* g4, const float* t4, int N) {
  __shared__ __align__(16) _Float16 sIn[WPB][16 * 192];
  __shared__ __align__(16) float    sOut[WPB][16 * 64];
  int wave = threadIdx.x >> 5, lane = threadIdx.x & 31;
  int tile = blockIdx.x * WPB + wave;
  int n0 = tile << 4;
  if (n0 >= N) return;
  _Float16* inT = sIn[wave];
  float*    outT = sOut[wave];

  for (int c = lane; c < 16 * 192; c += 32) {
    int row = c / 192, f = c - row * 192;
    int node = n0 + row; if (node >= N) node = N - 1;
    float v;
    if (f < 64)       v = mi[(node << 6) + f];
    else if (f < 128) v = mo[(node << 6) + f - 64];
    else              v = x [(node << 6) + f - 128];
    inT[row * 192 + f] = (_Float16)v;
  }
  wave_dsync();

  wmma_layer(inT, 192, 192, outT, W1, b1, lane);
  ln_relu_f16(outT, inT, 64, g1, t1, lane);
  wmma_layer(inT, 64, 64, outT, W2, b2, lane);
  ln_relu_f16(outT, inT, 64, g2, t2, lane);
  wmma_layer(inT, 64, 64, outT, W3, b3, lane);
  ln_relu_f16(outT, inT, 64, g3, t3, lane);
  wmma_layer(inT, 64, 64, outT, W4s, b4, lane);
  ln_relu_f16(outT, inT, 64, g4, t4, lane);

  int node = n0 + lane;
  if (lane < 16 && node < N) {               // residual: x = out + x0
    const _Float16* row = inT + (lane << 6);
    for (int j = 0; j < 64; ++j) {
      float nv = (float)row[j] + x[(node << 6) + j];
      x [(node << 6) + j] = nv;
      xh[(node << 6) + j] = (_Float16)nv;
    }
  }
}

// ---------------------------------------------------------------------------
// Node decoder: x(64) -> 64 -> 64 -> 64 (LN+ReLU) -> 1 + sigmoid
// ---------------------------------------------------------------------------
__global__ __launch_bounds__(TB) void node_dec_kernel(
    const _Float16* __restrict__ xh,
    const _Float16* W1, const _Float16* W2, const _Float16* W3,
    const float* b1, const float* g1, const float* t1,
    const float* b2, const float* g2, const float* t2,
    const float* b3, const float* g3, const float* t3,
    const float* W4, const float* b4,
    float* __restrict__ aout, int N) {
  __shared__ __align__(16) _Float16 sIn[WPB][16 * 64];
  __shared__ __align__(16) float    sOut[WPB][16 * 64];
  int wave = threadIdx.x >> 5, lane = threadIdx.x & 31;
  int tile = blockIdx.x * WPB + wave;
  int n0 = tile << 4;
  if (n0 >= N) return;
  _Float16* inT = sIn[wave];
  float*    outT = sOut[wave];

  for (int c = lane; c < 128; c += 32) {     // 16 rows x 8 chunks of 8 halves
    int row = c >> 3, part = c & 7;
    int node = n0 + row; if (node >= N) node = N - 1;
    *(uint4*)(inT + (row << 6) + (part << 3)) =
        *(const uint4*)(xh + (node << 6) + (part << 3));
  }
  wave_dsync();

  wmma_layer(inT, 64, 64, outT, W1, b1, lane);
  ln_relu_f16(outT, inT, 64, g1, t1, lane);
  wmma_layer(inT, 64, 64, outT, W2, b2, lane);
  ln_relu_f16(outT, inT, 64, g2, t2, lane);
  wmma_layer(inT, 64, 64, outT, W3, b3, lane);
  ln_relu_f16(outT, inT, 64, g3, t3, lane);

  int node = n0 + lane;
  if (lane < 16 && node < N) {
    float acc = b4[0];
    const _Float16* row = inT + (lane << 6);
    for (int j = 0; j < 64; ++j) acc += (float)row[j] * W4[j];
    aout[node] = 1.0f / (1.0f + expf(-acc));
  }
}

// ---------------------------------------------------------------------------
// Small helper kernels (memory-bound phases)
// ---------------------------------------------------------------------------
__global__ void zero_f32_kernel(float* p, int n) {
  int i = blockIdx.x * blockDim.x + threadIdx.x;
  if (i < n) p[i] = 0.f;
}

// f32 W[K][64] -> f16 fragment-swizzled (see load_B)
__global__ void swizzle_w_kernel(const float* __restrict__ W, _Float16* __restrict__ dst, int K) {
  int idx = blockIdx.x * blockDim.x + threadIdx.x;
  if (idx >= K * 64) return;
  int i = idx & 15;
  int h = (idx >> 4) & 1;
  int rest = idx >> 5;            // kb*64 + n
  int n = rest & 63;
  int kb = rest >> 6;
  int k = kb * 32 + h * 16 + i;
  dst[idx] = (_Float16)W[k * 64 + n];
}

// hit encoder: Linear(din->64) + LN + ReLU, atomic segment-sum into x[base+id]
__global__ void encode_hits_kernel(const float* __restrict__ hits, int din,
                                   const int* __restrict__ ids,
                                   const float* __restrict__ W, const float* __restrict__ b,
                                   const float* __restrict__ bt, const float* __restrict__ g,
                                   float* __restrict__ x, int base, int nHits) {
  int i = blockIdx.x * blockDim.x + threadIdx.x;
  if (i >= nHits) return;
  float in[8];
  for (int d = 0; d < din; ++d) in[d] = hits[i * din + d];
  float h[64]; float m = 0.f;
  for (int j = 0; j < 64; ++j) {
    float acc = b[j];
    for (int d = 0; d < din; ++d) acc += in[d] * W[d * 64 + j];
    h[j] = acc; m += acc;
  }
  m *= (1.0f / 64.0f);
  float var = 0.f;
  for (int j = 0; j < 64; ++j) { float d = h[j] - m; var += d * d; }
  var *= (1.0f / 64.0f);
  float inv = 1.0f / sqrtf(var + 1e-5f);
  int t = ids[i] + base;
  for (int j = 0; j < 64; ++j) {
    float y = (h[j] - m) * inv * g[j] + bt[j];
    atomicAdd(&x[t * 64 + j], y > 0.f ? y : 0.f);
  }
}

// track encoder: Linear(7->64)+LN+ReLU, x[row] += t, emit f16 mirror
__global__ void encode_tracks_kernel(const float* __restrict__ tr, int din,
                                     const float* __restrict__ W, const float* __restrict__ b,
                                     const float* __restrict__ bt, const float* __restrict__ g,
                                     float* __restrict__ x, _Float16* __restrict__ xh,
                                     int base, int nT) {
  int i = blockIdx.x * blockDim.x + threadIdx.x;
  if (i >= nT) return;
  float in[8];
  for (int d = 0; d < din; ++d) in[d] = tr[i * din + d];
  float h[64]; float m = 0.f;
  for (int j = 0; j < 64; ++j) {
    float acc = b[j];
    for (int d = 0; d < din; ++d) acc += in[d] * W[d * 64 + j];
    h[j] = acc; m += acc;
  }
  m *= (1.0f / 64.0f);
  float var = 0.f;
  for (int j = 0; j < 64; ++j) { float d = h[j] - m; var += d * d; }
  var *= (1.0f / 64.0f);
  float inv = 1.0f / sqrtf(var + 1e-5f);
  int row = base + i;
  for (int j = 0; j < 64; ++j) {
    float y = (h[j] - m) * inv * g[j] + bt[j];
    y = y > 0.f ? y : 0.f;
    float xv = x[row * 64 + j] + y;
    x[row * 64 + j] = xv;
    xh[row * 64 + j] = (_Float16)xv;
  }
}

// mi[n]=sum_v e[v,n-NVT]*x[v] (cdc nodes) ; mo[n]=sum_c e[n,c]*x[NVT+c] (vxd)
__global__ void aggregate_kernel(const float* __restrict__ e, const float* __restrict__ x,
                                 float* __restrict__ mi, float* __restrict__ mo,
                                 int NVT, int NCT) {
  int N = NVT + NCT;
  int idx = blockIdx.x * blockDim.x + threadIdx.x;
  if (idx >= N * 64) return;
  int n = idx >> 6, f = idx & 63;
  if (n < NVT) {
    float acc = 0.f;
    const float* er = e + n * NCT;
    for (int c = 0; c < NCT; ++c) acc += er[c] * x[((NVT + c) << 6) + f];
    mo[idx] = acc; mi[idx] = 0.f;
  } else {
    int c = n - NVT;
    float acc = 0.f;
    for (int v = 0; v < NVT; ++v) acc += e[v * NCT + c] * x[(v << 6) + f];
    mi[idx] = acc; mo[idx] = 0.f;
  }
}

// ---------------------------------------------------------------------------
extern "C" void kernel_launch(void* const* d_in, const int* in_sizes, int n_in,
                              void* d_out, int out_size, void* d_ws, size_t ws_size,
                              hipStream_t stream) {
  // Leaf order = jax sorted-key pytree flatten of setup_inputs():
  // params leaves first (keys sorted): edge_dec[16], edge_net[16], in_cdc_hits[4],
  // in_cdc_tracks[4], in_vxd_hits[4], in_vxd_tracks[4], node_dec[16], node_net[16]
  // (each layer: W, b, bt, g) -> 80 leaves; then vxd_hits, vxd_trackids,
  // vxd_tracks, cdc_hits, cdc_trackids, cdc_tracks (dict insertion order).
  const int ED = 0, EN = 16, ICH = 32, ICT = 36, IVH = 40, IVT = 44, ND = 48, NN = 64;
  const int DI = n_in - 6;  // first data input
  auto F = [&](int i) { return (const float*)d_in[i]; };

  int NVH = in_sizes[DI + 1];
  int NVT = in_sizes[DI + 2] / 7;
  int NCH = in_sizes[DI + 4];
  int NCT = in_sizes[DI + 5] / 7;
  int N = NVT + NCT;
  int E = NVT * NCT;

  char* ws = (char*)d_ws;
  size_t off = 0;
  auto alloc = [&](size_t bytes) {
    void* p = ws + off; off = (off + bytes + 255) & ~(size_t)255; return p;
  };
  float*    x    = (float*)alloc((size_t)N * 64 * 4);
  float*    mi   = (float*)alloc((size_t)N * 64 * 4);
  float*    mo   = (float*)alloc((size_t)N * 64 * 4);
  float*    ebuf = (float*)alloc((size_t)E * 4);
  _Float16* xh   = (_Float16*)alloc((size_t)N * 64 * 2);
  _Float16* enW1 = (_Float16*)alloc(128 * 64 * 2);
  _Float16* enW2 = (_Float16*)alloc(64 * 64 * 2);
  _Float16* enW3 = (_Float16*)alloc(64 * 64 * 2);
  _Float16* nnW1 = (_Float16*)alloc(192 * 64 * 2);
  _Float16* nnW2 = (_Float16*)alloc(64 * 64 * 2);
  _Float16* nnW3 = (_Float16*)alloc(64 * 64 * 2);
  _Float16* nnW4 = (_Float16*)alloc(64 * 64 * 2);
  _Float16* edW1 = (_Float16*)alloc(128 * 64 * 2);
  _Float16* edW2 = (_Float16*)alloc(64 * 64 * 2);
  _Float16* edW3 = (_Float16*)alloc(64 * 64 * 2);
  _Float16* ndW1 = (_Float16*)alloc(64 * 64 * 2);
  _Float16* ndW2 = (_Float16*)alloc(64 * 64 * 2);
  _Float16* ndW3 = (_Float16*)alloc(64 * 64 * 2);
  (void)ws_size;

  auto swz = [&](const float* W, _Float16* dst, int K) {
    swizzle_w_kernel<<<(K * 64 + 255) / 256, 256, 0, stream>>>(W, dst, K);
  };
  swz(F(EN + 0), enW1, 128); swz(F(EN + 4), enW2, 64); swz(F(EN + 8), enW3, 64);
  swz(F(NN + 0), nnW1, 192); swz(F(NN + 4), nnW2, 64); swz(F(NN + 8), nnW3, 64);
  swz(F(NN + 12), nnW4, 64);
  swz(F(ED + 0), edW1, 128); swz(F(ED + 4), edW2, 64); swz(F(ED + 8), edW3, 64);
  swz(F(ND + 0), ndW1, 64);  swz(F(ND + 4), ndW2, 64); swz(F(ND + 8), ndW3, 64);

  zero_f32_kernel<<<(N * 64 + 255) / 256, 256, 0, stream>>>(x, N * 64);

  encode_hits_kernel<<<(NVH + 127) / 128, 128, 0, stream>>>(
      F(DI + 0), 8, (const int*)d_in[DI + 1],
      F(IVH + 0), F(IVH + 1), F(IVH + 2), F(IVH + 3), x, 0, NVH);
  encode_hits_kernel<<<(NCH + 127) / 128, 128, 0, stream>>>(
      F(DI + 3), 5, (const int*)d_in[DI + 4],
      F(ICH + 0), F(ICH + 1), F(ICH + 2), F(ICH + 3), x, NVT, NCH);
  encode_tracks_kernel<<<(NVT + 127) / 128, 128, 0, stream>>>(
      F(DI + 2), 7, F(IVT + 0), F(IVT + 1), F(IVT + 2), F(IVT + 3), x, xh, 0, NVT);
  encode_tracks_kernel<<<(NCT + 127) / 128, 128, 0, stream>>>(
      F(DI + 5), 7, F(ICT + 0), F(ICT + 1), F(ICT + 2), F(ICT + 3), x, xh, NVT, NCT);

  int etiles = (E + 15) / 16, eblocks = (etiles + WPB - 1) / WPB;
  int ntiles = (N + 15) / 16, nblocks = (ntiles + WPB - 1) / WPB;

  for (int it = 0; it < 3; ++it) {
    edge_mlp_kernel<<<eblocks, TB, 0, stream>>>(
        xh, enW1, enW2, enW3,
        F(EN + 1), F(EN + 3), F(EN + 2),
        F(EN + 5), F(EN + 7), F(EN + 6),
        F(EN + 9), F(EN + 11), F(EN + 10),
        F(EN + 12), F(EN + 13), ebuf, NVT, NCT, E);
    aggregate_kernel<<<(N * 64 + 255) / 256, 256, 0, stream>>>(ebuf, x, mi, mo, NVT, NCT);
    node_mlp_kernel<<<nblocks, TB, 0, stream>>>(
        mi, mo, x, xh, nnW1, nnW2, nnW3, nnW4,
        F(NN + 1), F(NN + 3), F(NN + 2),
        F(NN + 5), F(NN + 7), F(NN + 6),
        F(NN + 9), F(NN + 11), F(NN + 10),
        F(NN + 13), F(NN + 15), F(NN + 14), N);
  }

  float* out = (float*)d_out;
  edge_mlp_kernel<<<eblocks, TB, 0, stream>>>(
      xh, edW1, edW2, edW3,
      F(ED + 1), F(ED + 3), F(ED + 2),
      F(ED + 5), F(ED + 7), F(ED + 6),
      F(ED + 9), F(ED + 11), F(ED + 10),
      F(ED + 12), F(ED + 13), out, NVT, NCT, E);
  node_dec_kernel<<<nblocks, TB, 0, stream>>>(
      xh, ndW1, ndW2, ndW3,
      F(ND + 1), F(ND + 3), F(ND + 2),
      F(ND + 5), F(ND + 7), F(ND + 6),
      F(ND + 9), F(ND + 11), F(ND + 10),
      F(ND + 12), F(ND + 13), out + E, N);
}
